// UnsupervisedMPNN_3753801416791
// MI455X (gfx1250) — matrix operand order repaired and checked by reading, not compiled, not run on hardware
//
#include <hip/hip_runtime.h>

#define H 32
#define MSG_WAVES 4
#define MSG_BLOCKS 400
#define GRU_BLOCKS 400

typedef __attribute__((ext_vector_type(16))) __bf16 v16bf;
typedef __attribute__((ext_vector_type(8)))  float  v8f;

union V16U { v16bf v; unsigned u[8]; uint4 q[2]; unsigned short h[16]; };

__device__ __forceinline__ unsigned pack2_bf16(float lo, float hi) {
  // result bytes = [lo.b2, lo.b3, hi.b2, hi.b3] -> two truncated bf16s
  return __builtin_amdgcn_perm(__float_as_uint(hi), __float_as_uint(lo), 0x07060302u);
}
__device__ __forceinline__ unsigned short trunc_bf16(float f) {
  return (unsigned short)(__float_as_uint(f) >> 16);
}
// CDNA5 VOP3P packed bf16 multiply (no clang builtin probed; ISA op 42).
__device__ __forceinline__ unsigned pk_mul_bf16(unsigned a, unsigned b) {
  unsigned d;
  asm("v_pk_mul_bf16 %0, %1, %2" : "=v"(d) : "v"(a), "v"(b));
  return d;
}
__device__ __forceinline__ v8f wmma_bf16(v16bf a, v16bf b, v8f c) {
  return __builtin_amdgcn_wmma_f32_16x16x32_bf16(false, a, false, b, (short)0, c,
                                                 false, false);
}

// A-fragment (16x32 bf16) from a 16-row tile of an fp32 row-major matrix.
// Lane L: row m=L&15; K runs {0..7,16..23} (L<16) or {8..15,24..31} (L>=16).
template <bool RELU>
__device__ __forceinline__ v16bf a_frag_from_f32(const float* base) {
  const int lane = threadIdx.x & 31;
  const int m = lane & 15;
  const int kofs = (lane < 16) ? 0 : 8;
  const float* p = base + m * H + kofs;
  float4 a0 = *(const float4*)(p);
  float4 a1 = *(const float4*)(p + 4);
  float4 b0 = *(const float4*)(p + 16);
  float4 b1 = *(const float4*)(p + 20);
  if (RELU) {
    a0.x = fmaxf(a0.x, 0.f); a0.y = fmaxf(a0.y, 0.f); a0.z = fmaxf(a0.z, 0.f); a0.w = fmaxf(a0.w, 0.f);
    a1.x = fmaxf(a1.x, 0.f); a1.y = fmaxf(a1.y, 0.f); a1.z = fmaxf(a1.z, 0.f); a1.w = fmaxf(a1.w, 0.f);
    b0.x = fmaxf(b0.x, 0.f); b0.y = fmaxf(b0.y, 0.f); b0.z = fmaxf(b0.z, 0.f); b0.w = fmaxf(b0.w, 0.f);
    b1.x = fmaxf(b1.x, 0.f); b1.y = fmaxf(b1.y, 0.f); b1.z = fmaxf(b1.z, 0.f); b1.w = fmaxf(b1.w, 0.f);
  }
  V16U r;
  r.u[0] = pack2_bf16(a0.x, a0.y); r.u[1] = pack2_bf16(a0.z, a0.w);
  r.u[2] = pack2_bf16(a1.x, a1.y); r.u[3] = pack2_bf16(a1.z, a1.w);
  r.u[4] = pack2_bf16(b0.x, b0.y); r.u[5] = pack2_bf16(b0.z, b0.w);
  r.u[6] = pack2_bf16(b1.x, b1.y); r.u[7] = pack2_bf16(b1.z, b1.w);
  return r.v;
}

// A-fragment from a 16-row tile of bf16 row-major matrix (tile-contiguous rows).
__device__ __forceinline__ v16bf a_frag_from_bf16(const unsigned short* base) {
  const int lane = threadIdx.x & 31;
  const int m = lane & 15;
  const int kofs = (lane < 16) ? 0 : 8;
  const unsigned short* p = base + m * H + kofs;
  V16U r;
  r.q[0] = *(const uint4*)(p);
  r.q[1] = *(const uint4*)(p + 16);
  return r.v;
}

// Gathered-row variant: caller supplies this lane's source row index.
__device__ __forceinline__ v16bf a_frag_from_bf16_row(const unsigned short* mat, int row) {
  const int lane = threadIdx.x & 31;
  const int kofs = (lane < 16) ? 0 : 8;
  const unsigned short* p = mat + (size_t)row * H + kofs;
  V16U r;
  r.q[0] = *(const uint4*)(p);
  r.q[1] = *(const uint4*)(p + 16);
  return r.v;
}

// B-fragment (32x16 bf16): 512 u16 per frag, lane L owns 16 contiguous bf16.
// Works for both global and LDS-resident fragment tables.
template <typename P>
__device__ __forceinline__ v16bf b_frag(const P* fragbase) {
  const int lane = threadIdx.x & 31;
  V16U r;
  r.q[0] = *(const uint4*)(fragbase + lane * 16);
  r.q[1] = *(const uint4*)(fragbase + lane * 16 + 8);
  return r.v;
}

// ---------------- weight fragment packing (run once per launch) -------------
// B frag element (lane L, t): K = base + (L>=16?8:0) + t + (t>=8?8:0), N = (L&15).
__global__ void k_prep(const float* en_w2, const float* en_b2, const float* lin0_w,
                       const float* en_w1, const float* w_ih, const float* w_hh,
                       unsigned short* w2frag, unsigned short* sfrag) {
  const int idx = blockIdx.x * blockDim.x + threadIdx.x;
  const int W2TOT = 66 * 512;  // 33 K-chunks x 2 N-halves
  if (idx < W2TOT) {
    int frag = idx >> 9, rr = idx & 511, L = rr >> 4, t = rr & 15;
    int c = frag >> 1, nh = frag & 1;
    int K = c * 32 + ((L >> 4) << 3) + t + ((t >> 3) << 3);   // row of W2'' (i*32+k, or 1024+i)
    int o = nh * 16 + (L & 15);             // output column
    float v;
    if (K < 1024) v = en_w2[(((K >> 5) << 5) + o) * 32 + (K & 31)];
    else          v = en_b2[((K - 1024) << 5) + o];
    w2frag[idx] = trunc_bf16(v);
  } else if (idx < W2TOT + 16 * 512) {
    int j = idx - W2TOT;
    int frag = j >> 9, rr = j & 511, L = rr >> 4, t = rr & 15;
    int k = ((L >> 4) << 3) + t + ((t >> 3) << 3);
    const float* Wm; int fl;
    if (frag < 2)       { Wm = lin0_w; fl = frag; }       // frags 0-1
    else if (frag < 4)  { Wm = en_w1;  fl = frag - 2; }   // frags 2-3
    else if (frag < 10) { Wm = w_ih;   fl = frag - 4; }   // frags 4-9
    else                { Wm = w_hh;   fl = frag - 10; }  // frags 10-15
    int n = fl * 16 + (L & 15);
    sfrag[j] = trunc_bf16(Wm[n * 32 + k]);   // B[k][n] = W[n,k]  (x @ W^T)
  }
}

// ---------------- generic 32->32 linear + bias + relu (lin0 / edge MLP) -----
template <bool WRITE_F32>
__global__ void k_in32out32(const float* X, const unsigned short* bfrags,
                            const float* bias, float* out_f32,
                            unsigned short* out_bf16, int n_rows) {
  const int lane = threadIdx.x & 31;
  const int widx = threadIdx.x >> 5;
  const int tile = blockIdx.x * (blockDim.x >> 5) + widx;
  if (tile >= (n_rows >> 4)) return;
  const int base = tile << 4;
  v16bf a  = a_frag_from_f32<false>(X + (size_t)base * H);
  v16bf b0 = b_frag(bfrags);
  v16bf b1 = b_frag(bfrags + 512);
  v8f z = {0.f, 0.f, 0.f, 0.f, 0.f, 0.f, 0.f, 0.f};
  v8f c0 = wmma_bf16(a, b0, z);
  v8f c1 = wmma_bf16(a, b1, z);
  const float bb0 = bias[lane & 15];
  const float bb1 = bias[16 + (lane & 15)];
#pragma unroll
  for (int r = 0; r < 8; ++r) {
    int m = (lane < 16) ? r : (r + 8);
    size_t row = (size_t)(base + m) * H;
    float v0 = fmaxf(c0[r] + bb0, 0.f);
    float v1 = fmaxf(c1[r] + bb1, 0.f);
    int col0 = lane & 15, col1 = 16 + (lane & 15);
    if (WRITE_F32) { out_f32[row + col0] = v0; out_f32[row + col1] = v1; }
    out_bf16[row + col0] = trunc_bf16(v0);
    out_bf16[row + col1] = trunc_bf16(v1);
  }
}

__global__ void k_zero(float* p, int n) {
  int i = blockIdx.x * blockDim.x + threadIdx.x;
  if (i < n) p[i] = 0.f;
}

// ---------------- fused message stage: msg = [outer(x,eh)|x] @ W2'' ---------
// 66 WMMAs per 16-edge subtile; A-fragment for K-chunk i is the eh fragment
// diag-scaled by x[m][i] via v_pk_mul_bf16 (8 VALU ops per fragment, no
// repack), so VALU ~ matches XDL issue and co-executes with WMMA. The 67.6 KB
// fragment table is staged in LDS once per block and shared by 4 waves that
// grid-stride over tiles (CDNA5: 320 KB LDS / WGP makes this cheap).
__global__ void k_msg(const unsigned short* h_bf, const unsigned short* eh_bf,
                      const unsigned short* w2frag, const int* src,
                      const int* dst, float* agg, int n_edges) {
  extern __shared__ char smem[];
  unsigned short* blds = (unsigned short*)smem;          // 66*512 u16
  unsigned* xs = (unsigned*)(smem + 66 * 512 * 2);       // 4*2*16*33 u32 (bf16 pairs)
  const int lane = threadIdx.x & 31;
  const int widx = threadIdx.x >> 5;
  const int mrow = lane & 15;

  // cooperative copy of the fragment table into LDS
  for (int i = threadIdx.x; i < 66 * 512 / 8; i += blockDim.x)
    ((uint4*)blds)[i] = ((const uint4*)w2frag)[i];
  __syncthreads();

  unsigned* myxs = xs + widx * (2 * 16 * 33);
  const int ntile32 = n_edges >> 5;
  const int nwaves = gridDim.x * MSG_WAVES;

  for (int t32 = blockIdx.x * MSG_WAVES + widx; t32 < ntile32; t32 += nwaves) {
    const int ebase = t32 << 5;
    v16bf xA[2];
    V16U ehA[2];
#pragma unroll
    for (int s = 0; s < 2; ++s) {
      const int sb = ebase + (s << 4);
      ehA[s].v = a_frag_from_bf16(eh_bf + (size_t)sb * H);
      int row = src[sb + mrow];
      xA[s] = a_frag_from_bf16_row(h_bf, row);
      for (int e = 0; e < 16; ++e) {
        int r2 = src[sb + e];
        unsigned hb = h_bf[(size_t)r2 * H + lane];   // bf16, zero-extended
        myxs[(s * 16 + e) * 33 + lane] = hb * 0x10001u;  // replicate to pair
      }
    }

    v8f z = {0.f, 0.f, 0.f, 0.f, 0.f, 0.f, 0.f, 0.f};
    v8f acc[2][2];
    acc[0][0] = z; acc[0][1] = z; acc[1][0] = z; acc[1][1] = z;

    for (int c = 0; c < 32; ++c) {
      v16bf b0 = b_frag(blds + (c * 2 + 0) * 512);
      v16bf b1 = b_frag(blds + (c * 2 + 1) * 512);
#pragma unroll
      for (int s = 0; s < 2; ++s) {
        unsigned scp = myxs[(s * 16 + mrow) * 33 + c];  // x[m][c] as bf16 pair
        V16U a;
#pragma unroll
        for (int j = 0; j < 8; ++j) a.u[j] = pk_mul_bf16(ehA[s].u[j], scp);
        acc[s][0] = wmma_bf16(a.v, b0, acc[s][0]);
        acc[s][1] = wmma_bf16(a.v, b1, acc[s][1]);
      }
    }
    { // bias rows of W2'' (K = 1024..1055): A = x itself
      v16bf b0 = b_frag(blds + 64 * 512);
      v16bf b1 = b_frag(blds + 65 * 512);
#pragma unroll
      for (int s = 0; s < 2; ++s) {
        acc[s][0] = wmma_bf16(xA[s], b0, acc[s][0]);
        acc[s][1] = wmma_bf16(xA[s], b1, acc[s][1]);
      }
    }
#pragma unroll
    for (int s = 0; s < 2; ++s) {
      const int sb = ebase + (s << 4);
#pragma unroll
      for (int r = 0; r < 8; ++r) {
        int m = (lane < 16) ? r : (r + 8);
        int d = dst[sb + m];
        float* ap = agg + (size_t)d * H + mrow;
        unsafeAtomicAdd(ap,      acc[s][0][r]);
        unsafeAtomicAdd(ap + 16, acc[s][1][r]);
      }
    }
  }
}

// ---------------- GRU: gi = relu(agg)@w_ih^T, gh = h@w_hh^T, gates, update --
__global__ void k_gru(const float* agg, float* hfp, unsigned short* h_bf,
                      const unsigned short* frag_ih, const unsigned short* frag_hh,
                      const float* b_ih, const float* b_hh, int n_nodes) {
  __shared__ unsigned short glds[12 * 512];  // 6 frags w_ih^T + 6 frags w_hh^T
  const int lane = threadIdx.x & 31;
  const int widx = threadIdx.x >> 5;

  for (int i = threadIdx.x; i < 6 * 512 / 8; i += blockDim.x) {
    ((uint4*)glds)[i]             = ((const uint4*)frag_ih)[i];
    ((uint4*)(glds + 6 * 512))[i] = ((const uint4*)frag_hh)[i];
  }
  __syncthreads();

  float bi[6], bh[6];
#pragma unroll
  for (int g = 0; g < 6; ++g) {
    bi[g] = b_ih[g * 16 + (lane & 15)];
    bh[g] = b_hh[g * 16 + (lane & 15)];
  }

  const int ntiles = n_nodes >> 4;
  const int nwaves = gridDim.x * (blockDim.x >> 5);
  for (int tile = blockIdx.x * (blockDim.x >> 5) + widx; tile < ntiles;
       tile += nwaves) {
    const int base = tile << 4;
    v16bf mA = a_frag_from_f32<true>(agg + (size_t)base * H);   // relu(agg)
    v16bf hA = a_frag_from_bf16(h_bf + (size_t)base * H);
    v8f z = {0.f, 0.f, 0.f, 0.f, 0.f, 0.f, 0.f, 0.f};
    v8f gi[6], gh[6];
#pragma unroll
    for (int g = 0; g < 6; ++g) {
      gi[g] = wmma_bf16(mA, b_frag(glds + g * 512), z);
      gh[g] = wmma_bf16(hA, b_frag(glds + (6 + g) * 512), z);
    }
#pragma unroll
    for (int r = 0; r < 8; ++r) {
      int m = (lane < 16) ? r : (r + 8);
      size_t row = (size_t)(base + m) * H;
#pragma unroll
      for (int p = 0; p < 2; ++p) {
        float rg = (gi[p][r] + bi[p]) + (gh[p][r] + bh[p]);
        float zg = (gi[2 + p][r] + bi[2 + p]) + (gh[2 + p][r] + bh[2 + p]);
        rg = 1.f / (1.f + __expf(-rg));
        zg = 1.f / (1.f + __expf(-zg));
        float ng = tanhf((gi[4 + p][r] + bi[4 + p]) + rg * (gh[4 + p][r] + bh[4 + p]));
        int col = (lane & 15) + (p << 4);
        float hold = hfp[row + col];
        float hn = (1.f - zg) * ng + zg * hold;
        hfp[row + col] = hn;
        h_bf[row + col] = trunc_bf16(hn);
      }
    }
  }
}

extern "C" void kernel_launch(void* const* d_in, const int* in_sizes, int n_in,
                              void* d_out, int out_size, void* d_ws, size_t ws_size,
                              hipStream_t stream) {
  const float* n_feat = (const float*)d_in[0];
  const float* e_feat = (const float*)d_in[1];
  const int*   src    = (const int*)d_in[2];
  const int*   dst    = (const int*)d_in[3];
  const float* lin0_w = (const float*)d_in[4];
  const float* lin0_b = (const float*)d_in[5];
  const float* en_w1  = (const float*)d_in[6];
  const float* en_b1  = (const float*)d_in[7];
  const float* en_w2  = (const float*)d_in[8];
  const float* en_b2  = (const float*)d_in[9];
  const float* w_ih   = (const float*)d_in[10];
  const float* w_hh   = (const float*)d_in[11];
  const float* b_ih   = (const float*)d_in[12];
  const float* b_hh   = (const float*)d_in[13];

  const int N = in_sizes[0] / H;   // 100000
  const int E = in_sizes[2];       // 200000
  float* h = (float*)d_out;        // h lives in d_out (fp32)

  unsigned short* h_bf   = (unsigned short*)d_ws;            // N*32 bf16
  unsigned short* eh_bf  = h_bf + (size_t)N * H;             // E*32 bf16
  float*          agg    = (float*)(eh_bf + (size_t)E * H);  // N*32 f32
  unsigned short* w2frag = (unsigned short*)(agg + (size_t)N * H);  // 66*512
  unsigned short* sfrag  = w2frag + 66 * 512;                       // 16*512
  const unsigned short* lin0_frag = sfrag;
  const unsigned short* enw1_frag = sfrag + 2 * 512;
  const unsigned short* ih_frag   = sfrag + 4 * 512;
  const unsigned short* hh_frag   = sfrag + 10 * 512;

  k_prep<<<(66 * 512 + 16 * 512 + 255) / 256, 256, 0, stream>>>(
      en_w2, en_b2, lin0_w, en_w1, w_ih, w_hh, w2frag, sfrag);

  // eh = relu(e_feat @ en_w1^T + en_b1) -> bf16 (hoisted, constant over steps)
  k_in32out32<false><<<(E / 16 + 3) / 4, 128, 0, stream>>>(
      e_feat, enw1_frag, en_b1, nullptr, eh_bf, E);
  // h0 = relu(n_feat @ lin0_w^T + lin0_b)
  k_in32out32<true><<<(N / 16 + 3) / 4, 128, 0, stream>>>(
      n_feat, lin0_frag, lin0_b, h, h_bf, N);

  const int msg_smem = 66 * 512 * 2 + MSG_WAVES * 2 * 16 * 33 * 4;  // 84480 B
  for (int step = 0; step < 6; ++step) {
    k_zero<<<(N * H + 255) / 256, 256, 0, stream>>>(agg, N * H);
    k_msg<<<MSG_BLOCKS, 32 * MSG_WAVES, msg_smem, stream>>>(
        h_bf, eh_bf, w2frag, src, dst, agg, E);
    k_gru<<<GRU_BLOCKS, 128, 0, stream>>>(agg, h, h_bf, ih_frag, hh_frag,
                                          b_ih, b_hh, N);
  }
  (void)n_in; (void)out_size; (void)ws_size;
}
